// SATT_74629351735385
// MI455X (gfx1250) — matrix operand order repaired and checked by reading, not compiled, not run on hardware
//
#include <hip/hip_runtime.h>

// ---------------------------------------------------------------------------
// Problem constants (match reference)
// ---------------------------------------------------------------------------
constexpr int BB   = 4;     // batch
constexpr int NN   = 512;   // nodes
constexpr int TT   = 12;    // time
constexpr int CIN  = 512;   // C_X + C_E
constexpr int COUT = 512;
constexpr int HH   = 8;     // heads
constexpr int DD   = 64;    // head dim

typedef __bf16 bf16;
typedef __bf16 v16bf __attribute__((ext_vector_type(16)));
typedef float  v8f   __attribute__((ext_vector_type(8)));

// ---------------------------------------------------------------------------
// CDNA5 async copy: GLOBAL_LOAD_ASYNC_TO_LDS_B128 (GVS form), ASYNCcnt-tracked.
// Per lane: LDS[ldsByte] = MEM[base + vOff], 16 bytes.  (ISA 08, VGLOBAL 98)
// ---------------------------------------------------------------------------
__device__ inline void async_copy_b128(unsigned int ldsByte, unsigned int vOff,
                                       const void* base) {
  asm volatile("global_load_async_to_lds_b128 %0, %1, %2"
               :: "v"(ldsByte), "v"(vOff), "s"(base)
               : "memory");
}
template <int N>
__device__ inline void wait_async() {
  asm volatile("s_wait_asynccnt %0" :: "n"(N) : "memory");
}
__device__ inline unsigned int lds_addr(const void* p) {
  // generic address of an LDS object: low 32 bits are the LDS byte offset
  return (unsigned int)(size_t)p;
}

// ---------------------------------------------------------------------------
// WMMA helpers (CDNA5 v_wmma_f32_16x16x32_bf16, VGPR layouts per ISA 7.12.2)
// ---------------------------------------------------------------------------
__device__ inline v8f wmma_bf16(v16bf a, v16bf b, v8f c) {
  return __builtin_amdgcn_wmma_f32_16x16x32_bf16(
      false, a, false, b, (short)0, c, false, false);
}

// A-matrix 16x32 fragment.  tile[row][k] row-major, leading dim ld.
__device__ inline v16bf load_a_frag(const bf16* tile, int lane, int mBase,
                                    int ld, int kBase) {
  int m    = mBase + (lane & 15);
  int kOff = (lane >> 4) ? 8 : 0;
  const bf16* p = tile + (size_t)m * ld + kBase + kOff;
  v16bf a;
#pragma unroll
  for (int j = 0; j < 8; ++j) a[j] = p[j];
#pragma unroll
  for (int j = 0; j < 8; ++j) a[8 + j] = p[16 + j];
  return a;
}

// B-matrix 32x16 fragment where B[k][col] == tile[col*ld + k].
__device__ inline v16bf load_b_frag(const bf16* tile, int lane, int colBase,
                                    int ld, int kBase) {
  int c    = colBase + (lane & 15);
  int kOff = (lane >> 4) ? 8 : 0;
  const bf16* p = tile + (size_t)c * ld + kBase + kOff;
  v16bf b;
#pragma unroll
  for (int j = 0; j < 8; ++j) b[j] = p[j];
#pragma unroll
  for (int j = 0; j < 8; ++j) b[8 + j] = p[16 + j];
  return b;
}

// ---------------------------------------------------------------------------
// Kernel 0a: concat(x, emb) + transpose [B,C,N,T] -> xcT[b][t][n][c] (bf16)
// ---------------------------------------------------------------------------
__global__ __launch_bounds__(256) void pack_x(const float* __restrict__ x,
                                              const float* __restrict__ spa,
                                              bf16* __restrict__ xcT) {
  size_t gid  = (size_t)blockIdx.x * blockDim.x + threadIdx.x;
  size_t idx0 = gid * 8;                       // total = B*T*N*CIN
  int c0 = idx0 & 511;
  int n  = (idx0 >> 9) & 511;
  int bt = (int)(idx0 >> 18);
  int t  = bt % TT, b = bt / TT;
  const float* src = (c0 < 256)
      ? (x   + (((size_t)b * 256 + c0)         * NN + n) * TT + t)
      : (spa + (((size_t)b * 256 + (c0 - 256)) * NN + n) * TT + t);
  union { bf16 h[8]; uint4 u; } pk;
#pragma unroll
  for (int j = 0; j < 8; ++j) pk.h[j] = (bf16)src[(size_t)j * NN * TT];
  *(uint4*)(xcT + idx0) = pk.u;
}

// Kernel 0b: convert 4 weight matrices [512x512] f32 -> bf16 (q,k,v,o)
__global__ __launch_bounds__(256) void pack_w(const float* __restrict__ Wq,
                                              const float* __restrict__ Wk,
                                              const float* __restrict__ Wv,
                                              const float* __restrict__ Wo,
                                              bf16* __restrict__ Wb) {
  size_t gid  = (size_t)blockIdx.x * blockDim.x + threadIdx.x;
  size_t idx0 = gid * 4;                       // total = 4*512*512
  int mat  = (int)(idx0 >> 18);
  size_t r = idx0 & 262143;
  const float* src = (mat == 0) ? Wq : (mat == 1) ? Wk : (mat == 2) ? Wv : Wo;
  float4 v = *(const float4*)(src + r);
  union { bf16 h[4]; uint2 u; } pk;
  pk.h[0] = (bf16)v.x; pk.h[1] = (bf16)v.y;
  pk.h[2] = (bf16)v.z; pk.h[3] = (bf16)v.w;
  *(uint2*)(Wb + idx0) = pk.u;
}

// ---------------------------------------------------------------------------
// Kernel 1: QKV projection (GEMM 64n x 64o per (b,t,mat), K in steps of 32).
// Async double-buffered LDS staging; q pre-scaled by 1/sqrt(D).
// Q,K stored [b][h][t][n][d];  V stored TRANSPOSED [b][h][t][d][n] so that
// the P*V B-fragments in the attention kernel are k-contiguous in LDS.
// ---------------------------------------------------------------------------
__global__ __launch_bounds__(128) void qkv_gemm(const bf16* __restrict__ xcT,
                                                const bf16* __restrict__ Wb,
                                                const float* __restrict__ bq,
                                                const float* __restrict__ bk,
                                                const float* __restrict__ bv,
                                                bf16* __restrict__ qkv) {
  __shared__ alignas(16) bf16 sA[2][64 * 32];
  __shared__ alignas(16) bf16 sB[2][64 * 32];
  const int tid = threadIdx.x, lane = tid & 31, w = tid >> 5;
  const int n0  = blockIdx.x * 64;
  const int mat = blockIdx.y >> 3;
  const int o0  = (blockIdx.y & 7) * 64;
  const int b   = blockIdx.z / TT, t = blockIdx.z % TT;

  const size_t aRow0 = ((size_t)(b * TT + t) * NN + n0) * CIN;   // elements
  const size_t bRow0 = ((size_t)mat * COUT + o0) * CIN;

  // tile = 64 rows x 32 k = 256 uint4; exactly 2 per thread per operand.
  auto issue = [&](int k0, int p) {
#pragma unroll
    for (int ii = 0; ii < 2; ++ii) {
      int u = tid + ii * 128;
      int r = u >> 2, c4 = u & 3;
      unsigned int gA = (unsigned int)((aRow0 + (size_t)r * CIN + k0) * 2) + c4 * 16;
      unsigned int gB = (unsigned int)((bRow0 + (size_t)r * CIN + k0) * 2) + c4 * 16;
      async_copy_b128(lds_addr(&sA[p][0]) + u * 16, gA, xcT);
      async_copy_b128(lds_addr(&sB[p][0]) + u * 16, gB, Wb);
    }
  };

  v8f acc[4];
#pragma unroll
  for (int s = 0; s < 4; ++s)
#pragma unroll
    for (int r = 0; r < 8; ++r) acc[s][r] = 0.f;

  issue(0, 0);
  for (int k0 = 0; k0 < CIN; k0 += 32) {
    int cur = (k0 >> 5) & 1;
    __syncthreads();                       // back buffer free to overwrite
    if (k0 + 32 < CIN) { issue(k0 + 32, cur ^ 1); wait_async<4>(); }
    else               { wait_async<0>(); }
    __syncthreads();                       // current tile visible to all waves

    v16bf a = load_a_frag(sA[cur], lane, w * 16, 32, 0);
#pragma unroll
    for (int s = 0; s < 4; ++s) {
      v16bf bf = load_b_frag(sB[cur], lane, s * 16, 32, 0);
      acc[s] = wmma_bf16(a, bf, acc[s]);
    }
  }

  const float* bias = (mat == 0) ? bq : (mat == 1) ? bk : bv;
  const float qscale = (mat == 0) ? 0.125f : 1.0f;
  bf16* dst = qkv + (size_t)mat * ((size_t)BB * HH * TT * NN * DD);
  const int rowHalf = (lane >> 4) ? 8 : 0;

  if (mat == 2) {
    // V: transposed store [b][h][t][d][n] -- 8 consecutive n per lane = b128
#pragma unroll
    for (int s = 0; s < 4; ++s) {
      int o = o0 + s * 16 + (lane & 15);
      int h = o >> 6, d = o & 63;
      float bia = bias[o];
      union { bf16 hh[8]; uint4 u; } pk;
#pragma unroll
      for (int r = 0; r < 8; ++r) pk.hh[r] = (bf16)(acc[s][r] + bia);
      bf16* base = dst + ((((size_t)b * HH + h) * TT + t) * NN) * DD +
                   (size_t)d * NN + n0 + w * 16 + rowHalf;
      *(uint4*)base = pk.u;
    }
  } else {
#pragma unroll
    for (int s = 0; s < 4; ++s) {
      int o = o0 + s * 16 + (lane & 15);
      int h = o >> 6, d = o & 63;
      float bia = bias[o];
      bf16* base = dst +
          (((((size_t)b * HH + h) * TT + t) * NN) + n0 + w * 16 + rowHalf) * DD + d;
#pragma unroll
      for (int r = 0; r < 8; ++r)
        base[(size_t)r * DD] = (bf16)((acc[s][r] + bia) * qscale);
    }
  }
}

// ---------------------------------------------------------------------------
// Kernel 2: flash attention per (b,h,t), 64 Q-rows per block (4 waves x 16).
// K/V tiles double-buffered via async-to-LDS; V panel is [d][n] so both the
// K and V B-fragments are k-contiguous (ds_load_b128 pairs).
// ---------------------------------------------------------------------------
__global__ __launch_bounds__(128) void attn_kernel(const bf16* __restrict__ qkv,
                                                   bf16* __restrict__ oT) {
  __shared__ alignas(16) bf16 sQ[64 * 64];
  __shared__ alignas(16) bf16 sK[2][64 * 64];      // [m][d]
  __shared__ alignas(16) bf16 sV[2][64 * 64];      // [d][m]  (transposed panel)
  __shared__ alignas(16) bf16 sP[4][16 * 64];      // wave-private P tiles

  const int tid = threadIdx.x, lane = tid & 31, w = tid >> 5;
  const int n0 = blockIdx.x * 64;
  const int h  = blockIdx.y / TT, t = blockIdx.y % TT;
  const int b  = blockIdx.z;
  const size_t matStride = (size_t)BB * HH * TT * NN * DD;   // elements
  const size_t head = ((((size_t)b * HH + h) * TT + t) * NN) * DD;

  const unsigned int kByte0 = (unsigned int)((matStride + head) * 2);
  const unsigned int vByte0 = (unsigned int)((2 * matStride + head) * 2);

  // K tile: 64 m-rows x 64 d (contiguous per panel);  V tile: 64 d-rows x 64 m
  auto issueKV = [&](int m0, int p) {
#pragma unroll
    for (int ii = 0; ii < 4; ++ii) {
      int u = tid + ii * 128;
      unsigned int l = u * 16;
      unsigned int gK = kByte0 + (unsigned int)(m0 * DD * 2) + u * 16;
      int r = u >> 3, c8 = u & 7;          // V: row=d, 8 uint4 per row of NN
      unsigned int gV = vByte0 +
          (unsigned int)(((size_t)r * NN + m0) * 2) + c8 * 16;
      async_copy_b128(lds_addr(&sK[p][0]) + l, gK, qkv);
      async_copy_b128(lds_addr(&sV[p][0]) + l, gV, qkv);
    }
  };

  // Q tile: contiguous 4096 bf16 -> plain vector copy
  {
    const uint4* src = (const uint4*)(qkv + head + (size_t)n0 * DD);
#pragma unroll
    for (int ii = 0; ii < 4; ++ii) ((uint4*)sQ)[tid + ii * 128] = src[tid + ii * 128];
  }

  const int rowHalf = (lane >> 4) ? 8 : 0;
  float mrow[8], lrow[8];
  v8f accO[4];
#pragma unroll
  for (int r = 0; r < 8; ++r) { mrow[r] = -3.0e38f; lrow[r] = 0.f; }
#pragma unroll
  for (int s = 0; s < 4; ++s)
#pragma unroll
    for (int r = 0; r < 8; ++r) accO[s][r] = 0.f;

  issueKV(0, 0);
  __syncthreads();
  v16bf aQ0 = load_a_frag(sQ, lane, w * 16, 64, 0);
  v16bf aQ1 = load_a_frag(sQ, lane, w * 16, 64, 32);

  for (int m0 = 0; m0 < NN; m0 += 64) {
    int cur = (m0 >> 6) & 1;
    __syncthreads();                       // back buffer free
    if (m0 + 64 < NN) { issueKV(m0 + 64, cur ^ 1); wait_async<8>(); }
    else              { wait_async<0>(); }
    __syncthreads();                       // tile visible block-wide

    // S = Q * K^T
    v8f S[4];
#pragma unroll
    for (int s = 0; s < 4; ++s) {
#pragma unroll
      for (int r = 0; r < 8; ++r) S[s][r] = 0.f;
      v16bf bk0 = load_b_frag(sK[cur], lane, s * 16, 64, 0);
      S[s] = wmma_bf16(aQ0, bk0, S[s]);
      v16bf bk1 = load_b_frag(sK[cur], lane, s * 16, 64, 32);
      S[s] = wmma_bf16(aQ1, bk1, S[s]);
    }

    // online softmax (rows live in 16-lane halves of the C-frag layout)
    float alpha[8];
#pragma unroll
    for (int r = 0; r < 8; ++r) {
      float mx = fmaxf(fmaxf(S[0][r], S[1][r]), fmaxf(S[2][r], S[3][r]));
      mx = fmaxf(mx, __shfl_xor(mx, 1, 32));
      mx = fmaxf(mx, __shfl_xor(mx, 2, 32));
      mx = fmaxf(mx, __shfl_xor(mx, 4, 32));
      mx = fmaxf(mx, __shfl_xor(mx, 8, 32));
      float mnew = fmaxf(mrow[r], mx);
      alpha[r] = __expf(mrow[r] - mnew);
      mrow[r] = mnew;
    }
    float psum[8];
#pragma unroll
    for (int r = 0; r < 8; ++r) psum[r] = 0.f;
#pragma unroll
    for (int s = 0; s < 4; ++s)
#pragma unroll
      for (int r = 0; r < 8; ++r) {
        float p = __expf(S[s][r] - mrow[r]);
        psum[r] += p;
        sP[w][(r + rowHalf) * 64 + s * 16 + (lane & 15)] = (bf16)p;
      }
#pragma unroll
    for (int r = 0; r < 8; ++r) {
      float rs = psum[r];
      rs += __shfl_xor(rs, 1, 32);
      rs += __shfl_xor(rs, 2, 32);
      rs += __shfl_xor(rs, 4, 32);
      rs += __shfl_xor(rs, 8, 32);
      lrow[r] = lrow[r] * alpha[r] + rs;
    }
#pragma unroll
    for (int s = 0; s < 4; ++s)
#pragma unroll
      for (int r = 0; r < 8; ++r) accO[s][r] *= alpha[r];

    // O += P * V   (B-frag: B[k=m][col=d] = sV[d*64 + m], k-contiguous)
#pragma unroll
    for (int ks = 0; ks < 2; ++ks) {
      v16bf aP = load_a_frag(sP[w], lane, 0, 64, ks * 32);
#pragma unroll
      for (int s = 0; s < 4; ++s) {
        v16bf bV = load_b_frag(sV[cur], lane, s * 16, 64, ks * 32);
        accO[s] = wmma_bf16(aP, bV, accO[s]);
      }
    }
  }

  // normalize + store bf16, head-major [n][d] layout
#pragma unroll
  for (int s = 0; s < 4; ++s) {
    int d = s * 16 + (lane & 15);
    bf16* base = oT + head + (size_t)(n0 + w * 16 + rowHalf) * DD + d;
#pragma unroll
    for (int r = 0; r < 8; ++r)
      base[(size_t)r * DD] = (bf16)(accO[s][r] / lrow[r]);
  }
}

// ---------------------------------------------------------------------------
// Kernel 3: output projection + bias + relu, scatter to [B,C,N,T] f32.
// ---------------------------------------------------------------------------
__global__ __launch_bounds__(128) void out_gemm(const bf16* __restrict__ oT,
                                                const bf16* __restrict__ Wb,
                                                const float* __restrict__ bo,
                                                float* __restrict__ out) {
  __shared__ alignas(16) bf16 sA[2][64 * 32];
  __shared__ alignas(16) bf16 sB[2][64 * 32];
  const int tid = threadIdx.x, lane = tid & 31, w = tid >> 5;
  const int n0 = blockIdx.x * 64;
  const int o0 = blockIdx.y * 64;
  const int b  = blockIdx.z / TT, t = blockIdx.z % TT;
  const size_t wOffElem = 3ull * COUT * CIN + (size_t)o0 * CIN;

  auto issue = [&](int k0, int p) {
    int hh = k0 >> 6, d0 = k0 & 63;   // 32-wide K slice stays inside one head
    size_t aRow0 = ((((size_t)b * HH + hh) * TT + t) * NN + n0) * DD + d0;
#pragma unroll
    for (int ii = 0; ii < 2; ++ii) {
      int u = tid + ii * 128;
      int r = u >> 2, c4 = u & 3;
      unsigned int gA = (unsigned int)((aRow0 + (size_t)r * DD) * 2) + c4 * 16;
      unsigned int gB = (unsigned int)((wOffElem + (size_t)r * CIN + k0) * 2) + c4 * 16;
      async_copy_b128(lds_addr(&sA[p][0]) + u * 16, gA, oT);
      async_copy_b128(lds_addr(&sB[p][0]) + u * 16, gB, Wb);
    }
  };

  v8f acc[4];
#pragma unroll
  for (int s = 0; s < 4; ++s)
#pragma unroll
    for (int r = 0; r < 8; ++r) acc[s][r] = 0.f;

  issue(0, 0);
  for (int k0 = 0; k0 < COUT; k0 += 32) {
    int cur = (k0 >> 5) & 1;
    __syncthreads();
    if (k0 + 32 < COUT) { issue(k0 + 32, cur ^ 1); wait_async<4>(); }
    else                { wait_async<0>(); }
    __syncthreads();

    v16bf a = load_a_frag(sA[cur], lane, w * 16, 32, 0);
#pragma unroll
    for (int s = 0; s < 4; ++s) {
      v16bf bf = load_b_frag(sB[cur], lane, s * 16, 32, 0);
      acc[s] = wmma_bf16(a, bf, acc[s]);
    }
  }

  const int rowHalf = (lane >> 4) ? 8 : 0;
#pragma unroll
  for (int s = 0; s < 4; ++s) {
    int o = o0 + s * 16 + (lane & 15);
    float bia = bo[o];
#pragma unroll
    for (int r = 0; r < 8; ++r) {
      int n = n0 + w * 16 + rowHalf + r;
      float v = fmaxf(acc[s][r] + bia, 0.f);
      out[(((size_t)b * COUT + o) * NN + n) * TT + t] = v;
    }
  }
}

// ---------------------------------------------------------------------------
// Launcher.  Workspace (bf16 elements): xcT | Wb(4 mats) | qkv(3 mats) | oT
// ---------------------------------------------------------------------------
extern "C" void kernel_launch(void* const* d_in, const int* in_sizes, int n_in,
                              void* d_out, int out_size, void* d_ws,
                              size_t ws_size, hipStream_t stream) {
  (void)in_sizes; (void)n_in; (void)out_size; (void)ws_size;
  const float* x   = (const float*)d_in[0];
  const float* spa = (const float*)d_in[1];
  const float* Wq  = (const float*)d_in[2];
  const float* bq  = (const float*)d_in[3];
  const float* Wk  = (const float*)d_in[4];
  const float* bk  = (const float*)d_in[5];
  const float* Wv  = (const float*)d_in[6];
  const float* bv  = (const float*)d_in[7];
  const float* Wo  = (const float*)d_in[8];
  const float* bo  = (const float*)d_in[9];
  float* out = (float*)d_out;

  bf16* xcT = (bf16*)d_ws;
  bf16* Wb  = xcT + (size_t)BB * TT * NN * CIN;
  bf16* qkv = Wb + 4ull * COUT * CIN;
  bf16* oT  = qkv + 3ull * BB * HH * TT * NN * DD;

  pack_x<<<6144, 256, 0, stream>>>(x, spa, xcT);
  pack_w<<<1024, 256, 0, stream>>>(Wq, Wk, Wv, Wo, Wb);
  qkv_gemm<<<dim3(8, 24, BB * TT), 128, 0, stream>>>(xcT, Wb, bq, bk, bv, qkv);
  attn_kernel<<<dim3(8, HH * TT, BB), 128, 0, stream>>>(qkv, oT);
  out_gemm<<<dim3(8, 8, BB * TT), 128, 0, stream>>>(oT, Wb, bo, out);
}
